// iTransformerEncoder_49667001811596
// MI455X (gfx1250) — matrix operand order repaired and checked by reading, not compiled, not run on hardware
//
#include <hip/hip_runtime.h>
#include <hip/hip_bf16.h>

// ---------------------------------------------------------------------------
// iTransformer encoder for MI455X (gfx1250):
//  - all GEMMs + attention matmuls on v_wmma_f32_16x16x32_f16 (wave32)
//  - f16 activations/weights feeding WMMA, f32 residual stream + LN/softmax
//  - DPP16 butterfly reductions (no ds_bpermute)
//  - TDM tensor_load_to_lds stages shared K/V chunks into LDS (TENSORcnt)
//    (6-arg builtin form: amdgpu-toolchain / clang-23)
// ---------------------------------------------------------------------------

typedef __attribute__((ext_vector_type(16))) _Float16 v16h;
typedef __attribute__((ext_vector_type(8)))  _Float16 v8h;
typedef __attribute__((ext_vector_type(8)))  float    v8f;
typedef __attribute__((ext_vector_type(4)))  unsigned int u32x4;
typedef __attribute__((ext_vector_type(8)))  int      i32x8;
typedef __attribute__((ext_vector_type(4)))  int      i32x4;

static constexpr int Bsz = 16, Sseq = 1024, Ttok = 512;
static constexpr int Dm = 256, Hh = 8, Ll = 4, FFd = 1024, DHd = 32;
static constexpr int Mrows = Bsz * Sseq;  // 16384

__device__ __forceinline__ v8f wmma_f16(v16h a, v16h b, v8f c) {
  return __builtin_amdgcn_wmma_f32_16x16x32_f16(false, a, false, b, (short)0, c,
                                                false, false);
}

__device__ __forceinline__ v16h cat16(v8h lo, v8h hi) {
  return __builtin_shufflevector(lo, hi, 0, 1, 2, 3, 4, 5, 6, 7, 8, 9, 10, 11,
                                 12, 13, 14, 15);
}

__device__ __forceinline__ v16h cvt16(float4 x0, float4 x1, float4 x2, float4 x3) {
  v16h r;
  r[0] = (_Float16)x0.x;  r[1] = (_Float16)x0.y;  r[2] = (_Float16)x0.z;  r[3] = (_Float16)x0.w;
  r[4] = (_Float16)x1.x;  r[5] = (_Float16)x1.y;  r[6] = (_Float16)x1.z;  r[7] = (_Float16)x1.w;
  r[8] = (_Float16)x2.x;  r[9] = (_Float16)x2.y;  r[10] = (_Float16)x2.z; r[11] = (_Float16)x2.w;
  r[12] = (_Float16)x3.x; r[13] = (_Float16)x3.y; r[14] = (_Float16)x3.z; r[15] = (_Float16)x3.w;
  return r;
}

// A-matrix 16x32 f16 tile: per-lane 8 halfs at kb and 8 at kb+16
// (kb = k0 + (lane>=16 ? 8 : 0)).
__device__ __forceinline__ v16h load_a(const float* row, int kb) {
  const float4* p0 = (const float4*)(row + kb);
  const float4* p1 = (const float4*)(row + kb + 16);
  return cvt16(p0[0], p0[1], p1[0], p1[1]);
}
__device__ __forceinline__ v16h load_a(const _Float16* row, int kb) {
  return cat16(*(const v8h*)(row + kb), *(const v8h*)(row + kb + 16));
}
// B-matrix 32x16 f16 tile: per-lane 8 halfs at kb and 8 at kb+16.
__device__ __forceinline__ v16h load_b16(const _Float16* p) {
  return cat16(*(const v8h*)p, *(const v8h*)(p + 16));
}

// DPP16 butterfly reductions within 16-lane rows (VALU, no LDS traffic).
// quad_perm(1,0,3,2)=0xB1, quad_perm(2,3,0,1)=0x4E,
// ROW_HALF_MIRROR=0x141, ROW_MIRROR=0x140.
template <int CTRL>
__device__ __forceinline__ float dpp_maxf(float x) {
  int y = __builtin_amdgcn_mov_dpp(__float_as_int(x), CTRL, 0xF, 0xF, true);
  return fmaxf(x, __int_as_float(y));
}
template <int CTRL>
__device__ __forceinline__ float dpp_addf(float x) {
  int y = __builtin_amdgcn_mov_dpp(__float_as_int(x), CTRL, 0xF, 0xF, true);
  return x + __int_as_float(y);
}
__device__ __forceinline__ float row16_max(float x) {
  x = dpp_maxf<0xB1>(x); x = dpp_maxf<0x4E>(x);
  x = dpp_maxf<0x141>(x); x = dpp_maxf<0x140>(x);
  return x;
}
__device__ __forceinline__ float row16_sum(float x) {
  x = dpp_addf<0xB1>(x); x = dpp_addf<0x4E>(x);
  x = dpp_addf<0x141>(x); x = dpp_addf<0x140>(x);
  return x;
}

// ---------------------------------------------------------------------------
// TDM: 2D tile (data_size = 2 bytes) global -> LDS. D# packing per
// CDNA5 ISA (group0) / §8.4 (group1).
// Builtin: 6-arg clang-23 form (g0, g1, g2, g3, extra, cpol).
// ---------------------------------------------------------------------------
__device__ __forceinline__ void tdm_load_2d_f16(uint32_t lds_addr,
                                                const void* gptr,
                                                uint32_t tile_d0,
                                                uint32_t tile_d1, uint32_t td0,
                                                uint32_t td1,
                                                uint64_t stride0_elems) {
  uint64_t ga = (uint64_t)(uintptr_t)gptr;
  u32x4 g0;
  g0[0] = 1u;                                  // count=1, user descriptor
  g0[1] = lds_addr;                            // lds_addr [63:32]
  g0[2] = (uint32_t)ga;                        // global_addr low
  g0[3] = (uint32_t)(ga >> 32) | (2u << 30);   // global_addr hi | type=2
  i32x8 g1;
  g1[0] = (int)(1u << 16);                     // data_size=1 (2B), mask=0
  g1[1] = (int)((td0 & 0xffffu) << 16);        // tensor_dim0[15:0]
  g1[2] = (int)((td0 >> 16) | ((td1 & 0xffffu) << 16));
  g1[3] = (int)((td1 >> 16) | (tile_d0 << 16));  // tile_dim0
  g1[4] = (int)tile_d1;                        // tile_dim1 (tile_dim2=0)
  g1[5] = (int)(uint32_t)(stride0_elems & 0xffffffffu);
  g1[6] = (int)(uint32_t)(stride0_elems >> 32);  // stride1=0
  g1[7] = 0;
  i32x4 gz = {0, 0, 0, 0};
  i32x8 gz8 = {0, 0, 0, 0, 0, 0, 0, 0};
  __builtin_amdgcn_tensor_load_to_lds(g0, g1, gz, gz, gz8, 0);
}

// ---------------------------------------------------------------------------
// Generic GEMM: Y[M,N] = act(X[M,K] @ W[N,K]^T + bias[N]) (+ R residual)
// wave -> 32x32 output tile, K-chunks of 32, 4 WMMAs/chunk, f16 W.
// ---------------------------------------------------------------------------
template <bool FUSE_RES, bool RELU, bool OUT_HALF, typename TA>
__global__ __launch_bounds__(256) void gemm_wmma_kernel(
    const TA* __restrict__ X, const _Float16* __restrict__ W,
    const float* __restrict__ bias, const float* __restrict__ R,
    void* __restrict__ Yv, int M, int N, int K) {
  const int wave = blockIdx.x * 8 + (threadIdx.x >> 5);
  const int lane = threadIdx.x & 31;
  const int ln15 = lane & 15;
  const int hf = lane >> 4;
  const int tiles_n = N >> 5;
  const int m0 = (wave / tiles_n) << 5;
  const int n0 = (wave % tiles_n) << 5;

  v8f c[2][2];
#pragma unroll
  for (int mi = 0; mi < 2; ++mi)
#pragma unroll
    for (int ni = 0; ni < 2; ++ni) {
      const int col = n0 + ni * 16 + ln15;
      const float bv = bias[col];
#pragma unroll
      for (int r = 0; r < 8; ++r) {
        float init = bv;
        if (FUSE_RES)
          init += R[(size_t)(m0 + mi * 16 + r + 8 * hf) * N + col];
        c[mi][ni][r] = init;
      }
    }

  const TA* xr0 = X + (size_t)(m0 + ln15) * K;
  const TA* xr1 = X + (size_t)(m0 + 16 + ln15) * K;
  const _Float16* wr0 = W + (size_t)(n0 + ln15) * K;
  const _Float16* wr1 = W + (size_t)(n0 + 16 + ln15) * K;
  const int akb = hf ? 8 : 0;
  const int bkb = hf ? 8 : 0;

  for (int k0 = 0; k0 < K; k0 += 32) {
    v16h a0 = load_a(xr0, k0 + akb);
    v16h a1 = load_a(xr1, k0 + akb);
    v16h b0 = load_b16(wr0 + k0 + bkb);
    v16h b1 = load_b16(wr1 + k0 + bkb);
    c[0][0] = wmma_f16(a0, b0, c[0][0]);
    c[0][1] = wmma_f16(a0, b1, c[0][1]);
    c[1][0] = wmma_f16(a1, b0, c[1][0]);
    c[1][1] = wmma_f16(a1, b1, c[1][1]);
  }

#pragma unroll
  for (int mi = 0; mi < 2; ++mi)
#pragma unroll
    for (int ni = 0; ni < 2; ++ni) {
      const int col = n0 + ni * 16 + ln15;
#pragma unroll
      for (int r = 0; r < 8; ++r) {
        float v = c[mi][ni][r];
        if (RELU) v = fmaxf(v, 0.0f);
        const size_t idx = (size_t)(m0 + mi * 16 + r + 8 * hf) * N + col;
        if (OUT_HALF)
          ((_Float16*)Yv)[idx] = (_Float16)v;
        else
          ((float*)Yv)[idx] = v;
      }
    }
}

// ---------------------------------------------------------------------------
// LayerNorm over D=256, one wave per row; DPP + single xor16 reduction.
// ---------------------------------------------------------------------------
__device__ __forceinline__ void store8(float* p, const float* o) {
  ((float4*)p)[0] = make_float4(o[0], o[1], o[2], o[3]);
  ((float4*)p)[1] = make_float4(o[4], o[5], o[6], o[7]);
}
__device__ __forceinline__ void store8(_Float16* p, const float* o) {
  v8h h;
#pragma unroll
  for (int j = 0; j < 8; ++j) h[j] = (_Float16)o[j];
  *(v8h*)p = h;
}

template <typename TOUT>
__global__ __launch_bounds__(256) void ln256_kernel(
    const float* __restrict__ in, const float* __restrict__ g,
    const float* __restrict__ b, TOUT* __restrict__ out, int M) {
  const int row = blockIdx.x * 8 + (threadIdx.x >> 5);
  const int lane = threadIdx.x & 31;
  const float* rp = in + (size_t)row * 256;
  float4 u0 = ((const float4*)rp)[lane * 2];
  float4 u1 = ((const float4*)rp)[lane * 2 + 1];
  float v[8] = {u0.x, u0.y, u0.z, u0.w, u1.x, u1.y, u1.z, u1.w};
  float s = 0.f;
#pragma unroll
  for (int j = 0; j < 8; ++j) s += v[j];
  s = row16_sum(s);
  s += __shfl_xor(s, 16, 32);
  const float mean = s * (1.0f / 256.0f);
  float q = 0.f;
#pragma unroll
  for (int j = 0; j < 8; ++j) { float d = v[j] - mean; q += d * d; }
  q = row16_sum(q);
  q += __shfl_xor(q, 16, 32);
  const float rstd = rsqrtf(q * (1.0f / 256.0f) + 1e-5f);
  float o[8];
#pragma unroll
  for (int j = 0; j < 8; ++j) {
    const int col = lane * 8 + j;
    o[j] = (v[j] - mean) * rstd * g[col] + b[col];
  }
  store8(out + (size_t)row * 256 + lane * 8, o);
}

// f32 -> f16 weight conversion (once per call; weights total ~6.5 MB).
__global__ __launch_bounds__(256) void cvtf16_kernel(
    const float* __restrict__ src, _Float16* __restrict__ dst, int n) {
  int i = blockIdx.x * 256 + threadIdx.x;
  if (i < n) dst[i] = (_Float16)src[i];
}

// ---------------------------------------------------------------------------
// RoPE: qkv f16 (M x 768) -> Qh/Kh f16 [b,h,s,32] (rope applied, f32 math).
// ---------------------------------------------------------------------------
__global__ __launch_bounds__(256) void rope_kernel(
    const _Float16* __restrict__ qkv, _Float16* __restrict__ Qh,
    _Float16* __restrict__ Kh) {
  const int t = blockIdx.x * 256 + threadIdx.x;  // B*H*S threads
  const int s = t & (Sseq - 1);
  const int hh = (t / Sseq) & (Hh - 1);
  const int b = t / (Sseq * Hh);
  const _Float16* qr = qkv + (size_t)(b * Sseq + s) * (3 * Dm) + hh * DHd;
  const _Float16* kr = qr + Dm;
  _Float16* qo = Qh + ((size_t)(b * Hh + hh) * Sseq + s) * DHd;
  _Float16* ko = Kh + ((size_t)(b * Hh + hh) * Sseq + s) * DHd;
#pragma unroll
  for (int d = 0; d < 16; ++d) {
    const float ang = (float)s * __powf(10000.0f, -(float)d * (1.0f / 16.0f));
    const float cs = __cosf(ang), sn = __sinf(ang);
    const float q0 = (float)qr[d], q1 = (float)qr[d + 16];
    const float k0 = (float)kr[d], k1 = (float)kr[d + 16];
    qo[d] = (_Float16)(q0 * cs - q1 * sn);
    qo[d + 16] = (_Float16)(q1 * cs + q0 * sn);
    ko[d] = (_Float16)(k0 * cs - k1 * sn);
    ko[d + 16] = (_Float16)(k1 * cs + k0 * sn);
  }
}

// V transpose: qkv v-part f16 -> Vt f16 [b,h,dh,S] (coalesced along s).
__global__ __launch_bounds__(256) void vtrans_kernel(
    const _Float16* __restrict__ qkv, _Float16* __restrict__ Vt) {
  const int t = blockIdx.x * 256 + threadIdx.x;  // B*H*32*S threads
  const int s = t & (Sseq - 1);
  const int d = (t / Sseq) & (DHd - 1);
  const int hh = (t / (Sseq * DHd)) & (Hh - 1);
  const int b = t / (Sseq * DHd * Hh);
  Vt[((size_t)(b * Hh + hh) * DHd + d) * Sseq + s] =
      qkv[(size_t)(b * Sseq + s) * (3 * Dm) + 2 * Dm + hh * DHd + d];
}

// ---------------------------------------------------------------------------
// Flash attention: block = 4 waves sharing one (b,h); each wave one 16-row
// Q tile.  Per 32-key chunk: wave0 TDM-stages K (1D 2KB) and V (2D 32x64B)
// into LDS -> QK^T (2 wmma) -> online softmax (DPP row reductions) ->
// P via LDS layout transform -> P·V (2 wmma).
// ---------------------------------------------------------------------------
__global__ __launch_bounds__(128) void attn_kernel(
    const _Float16* __restrict__ Qh, const _Float16* __restrict__ Kh,
    const _Float16* __restrict__ Vt, _Float16* __restrict__ O) {
  __shared__ __align__(16) _Float16 Kl[32][32];
  __shared__ __align__(16) _Float16 Vl[32][32];
  __shared__ __align__(16) _Float16 Pl[4][16][32];
  const int w = threadIdx.x >> 5;
  const int lane = threadIdx.x & 31;
  const int ln15 = lane & 15;
  const int hf = lane >> 4;
  const int bh = blockIdx.x >> 4;
  const int qblk = blockIdx.x & 15;
  const int q0 = qblk * 64 + w * 16;
  const int b = bh >> 3, hh = bh & 7;
  constexpr float SCALE = 0.17677669529663687f;  // 1/sqrt(32)

  const _Float16* Kbase = Kh + (size_t)bh * Sseq * DHd;
  const _Float16* Vbase = Vt + (size_t)bh * DHd * Sseq;
  const uint32_t ldsK = (uint32_t)(uintptr_t)(void*)&Kl[0][0];
  const uint32_t ldsV = (uint32_t)(uintptr_t)(void*)&Vl[0][0];

  // Q A-tile (16x32)
  const _Float16* qr =
      Qh + (size_t)bh * Sseq * DHd + (size_t)(q0 + ln15) * DHd + (hf ? 8 : 0);
  const v16h qa = cat16(*(const v8h*)qr, *(const v8h*)(qr + 16));

  float m_run[8], l_run[8];
  v8f o0, o1, vz;
#pragma unroll
  for (int r = 0; r < 8; ++r) {
    m_run[r] = -1e30f; l_run[r] = 0.f;
    o0[r] = 0.f; o1[r] = 0.f; vz[r] = 0.f;
  }

  for (int kc = 0; kc < Sseq; kc += 32) {
    __syncthreads();  // prior chunk's LDS reads complete before overwrite
    if (w == 0) {
      // K chunk: rows kc..kc+31 x 32 halfs = contiguous 2KB -> 1D tile.
      tdm_load_2d_f16(ldsK, Kbase + (size_t)kc * DHd, 1024, 0, 1024, 1, 1024);
      // V chunk: 32 dh-rows x 32 keys, row stride Sseq elements.
      tdm_load_2d_f16(ldsV, Vbase + kc, 32, 32, 1024, 32, (uint64_t)Sseq);
      __builtin_amdgcn_s_wait_tensorcnt(0);
    }
    __syncthreads();  // K/V staged for all 4 waves

    // K^T B-tiles from LDS: B[dh][key], contiguous dh per lane.
    v16h kb0 = load_b16(&Kl[ln15][hf ? 8 : 0]);
    v16h kb1 = load_b16(&Kl[16 + ln15][hf ? 8 : 0]);
    v8f s0 = wmma_f16(qa, kb0, vz);
    v8f s1 = wmma_f16(qa, kb1, vz);

#pragma unroll
    for (int r = 0; r < 8; ++r) {
      const float x0 = s0[r] * SCALE, x1 = s1[r] * SCALE;
      const float mx = row16_max(fmaxf(x0, x1));
      const float mnew = fmaxf(m_run[r], mx);
      const float fac = __expf(m_run[r] - mnew);
      const float p0 = __expf(x0 - mnew);
      const float p1 = __expf(x1 - mnew);
      const float rs = row16_sum(p0 + p1);
      l_run[r] = l_run[r] * fac + rs;
      m_run[r] = mnew;
      o0[r] *= fac;
      o1[r] *= fac;
      Pl[w][r + 8 * hf][ln15] = (_Float16)p0;
      Pl[w][r + 8 * hf][16 + ln15] = (_Float16)p1;
    }
    // this wave's P stores must land before its A-layout reads
    asm volatile("s_wait_dscnt 0" ::: "memory");

    const _Float16* pp = &Pl[w][ln15][hf ? 8 : 0];
    v16h pa = cat16(*(const v8h*)pp, *(const v8h*)(pp + 16));

    // V B-tiles from LDS: B[key][dh] via Vl[dh][key], contiguous keys.
    v16h vb0 = load_b16(&Vl[ln15][hf ? 8 : 0]);
    v16h vb1 = load_b16(&Vl[16 + ln15][hf ? 8 : 0]);
    o0 = wmma_f16(pa, vb0, o0);
    o1 = wmma_f16(pa, vb1, o1);
  }

#pragma unroll
  for (int r = 0; r < 8; ++r) {
    const float inv = 1.0f / l_run[r];
    const int srow = q0 + r + 8 * hf;
    _Float16* op = O + ((size_t)b * Sseq + srow) * Dm + hh * DHd;
    op[ln15] = (_Float16)(o0[r] * inv);
    op[16 + ln15] = (_Float16)(o1[r] * inv);
  }
}

// ---------------------------------------------------------------------------
// Host-side orchestration (all on `stream`, graph-capture safe).
// ---------------------------------------------------------------------------
extern "C" void kernel_launch(void* const* d_in, const int* in_sizes, int n_in,
                              void* d_out, int out_size, void* d_ws,
                              size_t ws_size, hipStream_t stream) {
  (void)in_sizes; (void)n_in; (void)out_size; (void)ws_size;
  const float* x         = (const float*)d_in[0];
  const float* tok_w     = (const float*)d_in[1];
  const float* tok_b     = (const float*)d_in[2];
  const float* tnorm_g   = (const float*)d_in[3];
  const float* tnorm_b   = (const float*)d_in[4];
  const float* in_proj_w = (const float*)d_in[5];
  const float* in_proj_b = (const float*)d_in[6];
  const float* out_w     = (const float*)d_in[7];
  const float* out_b     = (const float*)d_in[8];
  const float* ln1_g     = (const float*)d_in[9];
  const float* ln1_b     = (const float*)d_in[10];
  const float* ln2_g     = (const float*)d_in[11];
  const float* ln2_b     = (const float*)d_in[12];
  const float* lin1_w    = (const float*)d_in[13];
  const float* lin1_b    = (const float*)d_in[14];
  const float* lin2_w    = (const float*)d_in[15];
  const float* lin2_b    = (const float*)d_in[16];
  const float* fnorm_g   = (const float*)d_in[17];
  const float* fnorm_b   = (const float*)d_in[18];

  char* ws = (char*)d_ws;
  const size_t MB = (size_t)1 << 20;
  float*    hbuf = (float*)(ws);                  // 16 MB residual stream
  float*    tmp  = (float*)(ws + 16 * MB);        // 16 MB (tokenize out)
  _Float16* ybuf = (_Float16*)(ws + 32 * MB);     // 8 MB  LN out (GEMM A)
  _Float16* qkvh = (_Float16*)(ws + 40 * MB);     // 24 MB
  _Float16* atnh = (_Float16*)(ws + 64 * MB);     // 8 MB
  _Float16* zbuf = (_Float16*)(ws + 72 * MB);     // 32 MB FFN mid
  _Float16* Qh   = (_Float16*)(ws + 104 * MB);    // 8 MB
  _Float16* Kh   = (_Float16*)(ws + 112 * MB);    // 8 MB
  _Float16* Vt   = (_Float16*)(ws + 120 * MB);    // 8 MB
  _Float16* wtok = (_Float16*)(ws + 128 * MB);    // 256 KB
  _Float16* winp = (_Float16*)(ws + 129 * MB);    // 1.5 MB
  _Float16* wout = (_Float16*)(ws + 131 * MB);    // 512 KB
  _Float16* wl1  = (_Float16*)(ws + 132 * MB);    // 2 MB
  _Float16* wl2  = (_Float16*)(ws + 134 * MB);    // 2 MB

  const int M = Mrows;
  auto gblk = [](int Mm, int Nn) { return (Mm / 32) * (Nn / 32) / 8; };
  auto cvt = [&](const float* s, _Float16* d, int n) {
    cvtf16_kernel<<<(n + 255) / 256, 256, 0, stream>>>(s, d, n);
  };

  // weights -> f16 (once per call; deterministic)
  cvt(tok_w, wtok, Dm * Ttok);
  cvt(in_proj_w, winp, Ll * 3 * Dm * Dm);
  cvt(out_w, wout, Ll * Dm * Dm);
  cvt(lin1_w, wl1, Ll * FFd * Dm);
  cvt(lin2_w, wl2, Ll * Dm * FFd);

  // tokenize: tmp = x @ tok_w^T + tok_b ; h = LN(tmp)
  gemm_wmma_kernel<false, false, false><<<gblk(M, Dm), 256, 0, stream>>>(
      x, wtok, tok_b, (const float*)nullptr, (void*)tmp, M, Dm, Ttok);
  ln256_kernel<float><<<M / 8, 256, 0, stream>>>(tmp, tnorm_g, tnorm_b, hbuf, M);

  for (int i = 0; i < Ll; ++i) {
    // pre-norm attention
    ln256_kernel<_Float16><<<M / 8, 256, 0, stream>>>(hbuf, ln1_g + i * Dm,
                                                      ln1_b + i * Dm, ybuf, M);
    gemm_wmma_kernel<false, false, true><<<gblk(M, 3 * Dm), 256, 0, stream>>>(
        ybuf, winp + (size_t)i * 3 * Dm * Dm, in_proj_b + i * 3 * Dm,
        (const float*)nullptr, (void*)qkvh, M, 3 * Dm, Dm);
    rope_kernel<<<(Bsz * Hh * Sseq) / 256, 256, 0, stream>>>(qkvh, Qh, Kh);
    vtrans_kernel<<<(Bsz * Hh * DHd * Sseq) / 256, 256, 0, stream>>>(qkvh, Vt);
    attn_kernel<<<Bsz * Hh * (Sseq / 64), 128, 0, stream>>>(Qh, Kh, Vt, atnh);
    gemm_wmma_kernel<true, false, false><<<gblk(M, Dm), 256, 0, stream>>>(
        atnh, wout + (size_t)i * Dm * Dm, out_b + i * Dm, hbuf, (void*)hbuf,
        M, Dm, Dm);
    // pre-norm FFN
    ln256_kernel<_Float16><<<M / 8, 256, 0, stream>>>(hbuf, ln2_g + i * Dm,
                                                      ln2_b + i * Dm, ybuf, M);
    gemm_wmma_kernel<false, true, true><<<gblk(M, FFd), 256, 0, stream>>>(
        ybuf, wl1 + (size_t)i * FFd * Dm, lin1_b + i * FFd,
        (const float*)nullptr, (void*)zbuf, M, FFd, Dm);
    gemm_wmma_kernel<true, false, false><<<gblk(M, Dm), 256, 0, stream>>>(
        zbuf, wl2 + (size_t)i * Dm * FFd, lin2_b + i * Dm, hbuf, (void*)hbuf,
        M, Dm, FFd);
  }

  ln256_kernel<float><<<M / 8, 256, 0, stream>>>(hbuf, fnorm_g, fnorm_b,
                                                 (float*)d_out, M);
}
